// TritonGroupedExperts_18451179504156
// MI455X (gfx1250) — matrix operand order repaired and checked by reading, not compiled, not run on hardware
//
#include <hip/hip_runtime.h>
#include <hip/hip_bf16.h>
#include <math.h>

#define N_TOK   16384
#define D_MODEL 1024
#define D_FF    2048
#define N_EXP   8
#define TOP_K   2
#define PAIRS   (N_TOK * TOP_K)   // 32768

typedef __attribute__((ext_vector_type(16))) __bf16 v16bf;
typedef __attribute__((ext_vector_type(8)))  __bf16 v8bf;
typedef __attribute__((ext_vector_type(8)))  float  v8f;

union V16U { v16bf v; v8bf h[2]; };
union V8Z  { v8bf v; unsigned int u[4]; };

static __device__ __forceinline__ v8f wmma_bf16(v16bf a, v16bf b, v8f c) {
  // v_wmma_f32_16x16x32_bf16: D = A(16x32) * B(32x16) + C
  return __builtin_amdgcn_wmma_f32_16x16x32_bf16(false, a, false, b, (short)0, c,
                                                 false, false);
}

static __device__ __forceinline__ v8f vzero8f() {
  v8f z = {0.f,0.f,0.f,0.f,0.f,0.f,0.f,0.f};
  return z;
}

// ---------------- routing ----------------

__global__ void moe_count(const int* __restrict__ idx, int* __restrict__ cnt) {
  int i = blockIdx.x * blockDim.x + threadIdx.x;
  if (i < PAIRS) atomicAdd(&cnt[idx[i]], 1);
}

__global__ void moe_scan(const int* __restrict__ cnt, int* __restrict__ off,
                         int* __restrict__ cur) {
  if (threadIdx.x == 0) {
    int s = 0;
    for (int e = 0; e < N_EXP; ++e) { off[e] = s; cur[e] = s; s += cnt[e]; }
  }
}

__global__ void moe_scatter(const int* __restrict__ idx, const float* __restrict__ wts,
                            int* __restrict__ cur, int* __restrict__ tok,
                            float* __restrict__ tw) {
  int i = blockIdx.x * blockDim.x + threadIdx.x;
  if (i < PAIRS) {
    int e = idx[i];
    int p = atomicAdd(&cur[e], 1);
    tok[p] = i >> 1;          // token id (pair i = token*TOP_K + k)
    tw[p]  = wts[i];          // routing coefficient for this slot
  }
}

// ---------------- FFN stage 1: H = coef * silu(x@w1) * (x@w2)  (grouped) ----------------
// block: 256 thr = 8 waves (2x4). Tile: 64 pairs x 128 ff cols, K-step 32 over d_model.

__global__ __launch_bounds__(256) void moe_ffn1(
    const float* __restrict__ x, const float* __restrict__ w1,
    const float* __restrict__ w2, const int* __restrict__ cnt,
    const int* __restrict__ off, const int* __restrict__ tok,
    const float* __restrict__ tw, __bf16* __restrict__ H)
{
  const int e    = blockIdx.y;
  const int mt   = blockIdx.x;
  const int cntE = cnt[e];
  if (mt * 64 >= cntE) return;                 // uniform early-exit
  const int offE  = off[e];
  const int fBase = blockIdx.z * 128;

  __shared__ __attribute__((aligned(32))) __bf16 As [64][32];
  __shared__ __attribute__((aligned(32))) __bf16 B1s[128][32];   // [f][k] (K-major)
  __shared__ __attribute__((aligned(32))) __bf16 B2s[128][32];
  __shared__ float coefs[64];
  __shared__ int   toks [64];

  const int tid = threadIdx.x;
  if (tid < 64) {
    bool v = (mt * 64 + tid) < cntE;
    toks[tid]  = v ? tok[offE + mt * 64 + tid] : -1;
    coefs[tid] = v ? tw [offE + mt * 64 + tid] : 0.f;
  }
  __syncthreads();

  const int lane = tid & 31;
  const int wv   = tid >> 5;
  const int wm   = wv >> 2;                    // 0..1  (32 rows each)
  const int wn   = wv & 3;                     // 0..3  (32 cols each)

  v8f acc1[2][2], acc2[2][2];
#pragma unroll
  for (int i = 0; i < 2; ++i)
#pragma unroll
    for (int j = 0; j < 2; ++j) { acc1[i][j] = vzero8f(); acc2[i][j] = vzero8f(); }

  // per-thread staging assignments
  const int ar = tid >> 2;                     // A row 0..63
  const int ac = (tid & 3) * 8;                // A col seg
  const int tA = toks[ar];
  const float* xrow = (tA >= 0) ? (x + (size_t)tA * D_MODEL) : nullptr;
  const int bk = tid >> 3;                     // w k-row 0..31
  const int bf = (tid & 7) * 16;               // w f seg
  const float* w1e = w1 + (size_t)e * D_MODEL * D_FF;
  const float* w2e = w2 + (size_t)e * D_MODEL * D_FF;

  for (int kk = 0; kk < D_MODEL; kk += 32) {
    // stage A (fp32 -> bf16)
    if (xrow) {
#pragma unroll
      for (int j = 0; j < 8; ++j) As[ar][ac + j] = (__bf16)xrow[kk + ac + j];
    } else {
#pragma unroll
      for (int j = 0; j < 8; ++j) As[ar][ac + j] = (__bf16)0.f;
    }
    // stage B1/B2 transposed to K-major
    const float* p1 = w1e + (size_t)(kk + bk) * D_FF + fBase + bf;
    const float* p2 = w2e + (size_t)(kk + bk) * D_FF + fBase + bf;
#pragma unroll
    for (int j = 0; j < 16; ++j) {
      B1s[bf + j][bk] = (__bf16)p1[j];
      B2s[bf + j][bk] = (__bf16)p2[j];
    }
    __syncthreads();

    // A fragments (16x32 bf16): lane<16 -> K 0-15 ; lane>=16 -> K 16-31
    v16bf a[2], b1[2], b2[2];
    const int kbA = (lane < 16) ? 0 : 16;
#pragma unroll
    for (int mf = 0; mf < 2; ++mf) {
      const int row = wm * 32 + mf * 16 + (lane & 15);
      V16U u;
      u.h[0] = *(const v8bf*)&As[row][kbA];
      u.h[1] = *(const v8bf*)&As[row][kbA + 8];
      a[mf] = u.v;
    }
    // B fragments (32x16): lane column, 16 contiguous K
    const int kbB = (lane < 16) ? 0 : 16;
#pragma unroll
    for (int nf = 0; nf < 2; ++nf) {
      const int col = wn * 32 + nf * 16 + (lane & 15);
      b1[nf] = *(const v16bf*)&B1s[col][kbB];
      b2[nf] = *(const v16bf*)&B2s[col][kbB];
    }
#pragma unroll
    for (int mf = 0; mf < 2; ++mf)
#pragma unroll
      for (int nf = 0; nf < 2; ++nf) {
        acc1[mf][nf] = wmma_bf16(a[mf], b1[nf], acc1[mf][nf]);
        acc2[mf][nf] = wmma_bf16(a[mf], b2[nf], acc2[mf][nf]);
      }
    __syncthreads();
  }

  // epilogue: H = coef * silu(gate) * up   (C layout: vgpr j -> M=j / j+8)
#pragma unroll
  for (int mf = 0; mf < 2; ++mf) {
    const int mB = wm * 32 + mf * 16;
#pragma unroll
    for (int nf = 0; nf < 2; ++nf) {
      const int nB = fBase + wn * 32 + nf * 16 + (lane & 15);
#pragma unroll
      for (int j = 0; j < 8; ++j) {
        const int row = mB + ((lane < 16) ? j : j + 8);
        if (mt * 64 + row < cntE) {
          float g = acc1[mf][nf][j];
          float u = acc2[mf][nf][j];
          float h = coefs[row] * (g / (1.f + __expf(-g))) * u;
          H[(size_t)(offE + mt * 64 + row) * D_FF + nB] = (__bf16)h;
        }
      }
    }
  }
}

// ---------------- FFN stage 2: out[t] += H[p] @ w3[e]  (grouped, atomic combine) ----------------

__global__ __launch_bounds__(256) void moe_ffn2(
    const __bf16* __restrict__ H, const float* __restrict__ w3,
    const int* __restrict__ cnt, const int* __restrict__ off,
    const int* __restrict__ tok, float* __restrict__ out)
{
  const int e    = blockIdx.y;
  const int mt   = blockIdx.x;
  const int cntE = cnt[e];
  if (mt * 64 >= cntE) return;
  const int offE  = off[e];
  const int nBase = blockIdx.z * 128;

  __shared__ __attribute__((aligned(32))) __bf16 As[64][32];
  __shared__ __attribute__((aligned(32))) __bf16 Bs[128][32];
  __shared__ int toks[64];

  const int tid = threadIdx.x;
  if (tid < 64) {
    bool v = (mt * 64 + tid) < cntE;
    toks[tid] = v ? tok[offE + mt * 64 + tid] : -1;
  }
  __syncthreads();

  const int lane = tid & 31;
  const int wv   = tid >> 5;
  const int wm   = wv >> 2;
  const int wn   = wv & 3;

  v8f acc[2][2];
#pragma unroll
  for (int i = 0; i < 2; ++i)
#pragma unroll
    for (int j = 0; j < 2; ++j) acc[i][j] = vzero8f();

  const int ar = tid >> 2;
  const int ac = (tid & 3) * 8;
  const bool arv = (mt * 64 + ar) < cntE;
  const __bf16* hrow = H + (size_t)(offE + mt * 64 + ar) * D_FF;
  const int bk = tid >> 3;
  const int bf = (tid & 7) * 16;
  const float* w3e = w3 + (size_t)e * D_FF * D_MODEL;

  for (int kk = 0; kk < D_FF; kk += 32) {
    if (arv) {
      *(v8bf*)&As[ar][ac] = *(const v8bf*)(hrow + kk + ac);   // bf16 already
    } else {
      V8Z z; z.u[0] = z.u[1] = z.u[2] = z.u[3] = 0u;
      *(v8bf*)&As[ar][ac] = z.v;
    }
    const float* p3 = w3e + (size_t)(kk + bk) * D_MODEL + nBase + bf;
#pragma unroll
    for (int j = 0; j < 16; ++j) Bs[bf + j][bk] = (__bf16)p3[j];
    __syncthreads();

    v16bf a[2], b[2];
    const int kbA = (lane < 16) ? 0 : 16;
#pragma unroll
    for (int mf = 0; mf < 2; ++mf) {
      const int row = wm * 32 + mf * 16 + (lane & 15);
      V16U u;
      u.h[0] = *(const v8bf*)&As[row][kbA];
      u.h[1] = *(const v8bf*)&As[row][kbA + 8];
      a[mf] = u.v;
    }
    const int kbB = (lane < 16) ? 0 : 16;
#pragma unroll
    for (int nf = 0; nf < 2; ++nf) {
      const int col = wn * 32 + nf * 16 + (lane & 15);
      b[nf] = *(const v16bf*)&Bs[col][kbB];
    }
#pragma unroll
    for (int mf = 0; mf < 2; ++mf)
#pragma unroll
      for (int nf = 0; nf < 2; ++nf)
        acc[mf][nf] = wmma_bf16(a[mf], b[nf], acc[mf][nf]);
    __syncthreads();
  }

#pragma unroll
  for (int mf = 0; mf < 2; ++mf) {
    const int mB = wm * 32 + mf * 16;
#pragma unroll
    for (int nf = 0; nf < 2; ++nf) {
      const int col = nBase + wn * 32 + nf * 16 + (lane & 15);
#pragma unroll
      for (int j = 0; j < 8; ++j) {
        const int row = mB + ((lane < 16) ? j : j + 8);
        const int t   = toks[row];
        if (t >= 0) {
          unsafeAtomicAdd(&out[(size_t)t * D_MODEL + col], acc[mf][nf][j]);
        }
      }
    }
  }
}

// ---------------- launch ----------------
// ws layout: [0] cnt(8 int) [32] off(8 int) [64] cur(8 int)
//            [128] tok(32768 int) [+128KB] tw(32768 f32)
//            [512KB] H (32768 x 2048 bf16 = 128 MB)  -> total ~135 MB

extern "C" void kernel_launch(void* const* d_in, const int* in_sizes, int n_in,
                              void* d_out, int out_size, void* d_ws, size_t ws_size,
                              hipStream_t stream) {
  const float* x    = (const float*)d_in[0];
  const int*   eidx = (const int*)  d_in[1];
  const float* ew   = (const float*)d_in[2];
  const float* w1   = (const float*)d_in[3];
  const float* w2   = (const float*)d_in[4];
  const float* w3   = (const float*)d_in[5];
  float* out = (float*)d_out;

  char* ws  = (char*)d_ws;
  int*   cnt = (int*)(ws + 0);
  int*   off = (int*)(ws + 32);
  int*   cur = (int*)(ws + 64);
  int*   tok = (int*)(ws + 128);
  float* tw  = (float*)(ws + 128 + (size_t)PAIRS * 4);
  __bf16* H  = (__bf16*)(ws + (1u << 19));

  hipMemsetAsync(d_out, 0, (size_t)N_TOK * D_MODEL * sizeof(float), stream);
  hipMemsetAsync(cnt, 0, 32, stream);

  moe_count  <<<PAIRS / 256, 256, 0, stream>>>(eidx, cnt);
  moe_scan   <<<1, 32, 0, stream>>>(cnt, off, cur);
  moe_scatter<<<PAIRS / 256, 256, 0, stream>>>(eidx, ew, cur, tok, tw);

  dim3 g1(PAIRS / 64, N_EXP, D_FF / 128);     // (512, 8, 16), early-exit past bucket end
  moe_ffn1<<<g1, 256, 0, stream>>>(x, w1, w2, cnt, off, tok, tw, H);

  dim3 g2(PAIRS / 64, N_EXP, D_MODEL / 128);  // (512, 8, 8)
  moe_ffn2<<<g2, 256, 0, stream>>>(H, w3, cnt, off, tok, out);
}
